// AtlasMAGSkeleton_30434138259496
// MI455X (gfx1250) — matrix operand order repaired and checked by reading, not compiled, not run on hardware
//
#include <hip/hip_runtime.h>
#include <hip/hip_bf16.h>

// ---------------------------------------------------------------------------
// AtlasMAG forward for MI455X (gfx1250, wave32, WMMA).
// - All GEMMs + both attention matmul passes use v_wmma_f32_16x16x32_f16.
// - Weights/activations cached as f16 once per launch (halves HBM traffic).
// - GEMM fast path: double-buffered GLOBAL_LOAD_ASYNC_TO_LDS_B128 staging
//   with s_wait_asynccnt (CDNA5 async-copy path), guarded by __has_builtin.
// ---------------------------------------------------------------------------

typedef __attribute__((ext_vector_type(16))) _Float16 v16h;
typedef __attribute__((ext_vector_type(8)))  _Float16 v8h;
typedef __attribute__((ext_vector_type(8)))  float    v8f;
typedef int gv4i __attribute__((vector_size(16)));  // matches builtin param type

#define KB  2          // batch
#define KS  1024       // seq
#define KD  1024       // dim
#define KH  16         // heads
#define KDH 64         // head dim
#define KL  6          // layers
#define KV  32000      // vocab
#define KNP 16         // persistent tokens
#define KGH 64         // gate hidden
#define KFF 2730       // swiglu hidden
#define KFFP 2752      // KFF padded to multiple of 32 (16B-aligned f16 rows)

#if __has_builtin(__builtin_amdgcn_global_load_async_to_lds_b128)
#define ASYNC_COPY 1
#else
#define ASYNC_COPY 0
#endif

__device__ inline v8f wmma16(v16h a, v16h b, v8f c) {
  return __builtin_amdgcn_wmma_f32_16x16x32_f16(false, a, false, b, (short)0, c,
                                                false, false);
}

#if ASYNC_COPY
__device__ inline void async_cp16(const _Float16* g, _Float16* l) {
  __builtin_amdgcn_global_load_async_to_lds_b128((gv4i*)g, (gv4i*)l, 0, 0);
}
__device__ inline void wait_async0() {
#if __has_builtin(__builtin_amdgcn_s_wait_asynccnt)
  __builtin_amdgcn_s_wait_asynccnt(0);
#else
  asm volatile("s_wait_asynccnt 0" ::: "memory");
#endif
}
#endif

// A-operand fragment (16xK tile, K=32 slice). Contiguous k layout (matches B):
// lane<16:  elems 0..15 = K0..15  (row M = lane)
// lane>=16: elems 0..15 = K16..31 (row M = lane-16)
__device__ inline v16h load_a_frag(const _Float16* base, int ld) {
  int l = threadIdx.x & 31;
  const _Float16* p = base + (size_t)(l & 15) * ld + ((l >> 4) * 16);
  v8h lo = *(const v8h*)(p);
  v8h hi = *(const v8h*)(p + 8);
  v16h r;
#pragma unroll
  for (int j = 0; j < 8; ++j) { r[j] = lo[j]; r[8 + j] = hi[j]; }
  return r;
}

// B-operand fragment (Kx16 tile, K=32). element (k, n) at base[n*ld + k].
__device__ inline v16h load_b_frag(const _Float16* base, int ld) {
  int l = threadIdx.x & 31;
  return *(const v16h*)(base + (size_t)(l & 15) * ld + ((l >> 4) * 16));
}

__device__ inline v8f zero8() { v8f z = {}; return z; }

// ---------------------------------------------------------------------------
// fp32 -> f16 weight caching
// ---------------------------------------------------------------------------
__global__ void cvt16_kernel(const float* __restrict__ src,
                             _Float16* __restrict__ dst, long long n) {
  long long i = (long long)blockIdx.x * 256 + threadIdx.x;
  if (i < n) dst[i] = (_Float16)src[i];
}

// ffn_w3 (L, KD, KFF) -> f16 padded to (L, KD, KFFP), zero fill
__global__ void cvtw3_kernel(const float* __restrict__ src,
                             _Float16* __restrict__ dst) {
  long long i = (long long)blockIdx.x * 256 + threadIdx.x;
  long long n = (long long)KL * KD * KFFP;
  if (i >= n) return;
  int c = (int)(i % KFFP);
  long long row = i / KFFP;  // l*KD + r
  dst[i] = (c < KFF) ? (_Float16)src[row * KFF + c] : (_Float16)0.f;
}

// ---------------------------------------------------------------------------
// Embedding: x[bs][d] = tok_emb[ids[bs]][d] + w_init[d]
// ---------------------------------------------------------------------------
__global__ void embed_kernel(const int* __restrict__ ids,
                             const float* __restrict__ emb,
                             const float* __restrict__ w_init,
                             float* __restrict__ x) {
  int bs = blockIdx.x;
  const float* e = emb + (size_t)ids[bs] * KD;
  float* xo = x + (size_t)bs * KD;
  for (int d = threadIdx.x; d < KD; d += 256) xo[d] = e[d] + w_init[d];
}

// ---------------------------------------------------------------------------
// RoPE tables: cos/sin (S, 32)
// ---------------------------------------------------------------------------
__global__ void rope_kernel(float* __restrict__ cosb, float* __restrict__ sinb) {
  int idx = blockIdx.x * 256 + threadIdx.x;
  if (idx >= KS * 32) return;
  int f = idx & 31, s = idx >> 5;
  float inv = powf(10000.f, -(2.f * (float)f) / (float)KDH);
  float fr = (float)s * inv;
  cosb[idx] = cosf(fr);
  sinb[idx] = sinf(fr);
}

// ---------------------------------------------------------------------------
// Persistent memory: Mp[h][d][e] = sum_n p[n][h][d]*p[n][h][e]  (f16 out)
// ---------------------------------------------------------------------------
__global__ void mp_kernel(const float* __restrict__ pers, _Float16* __restrict__ mp16) {
  int idx = blockIdx.x * 256 + threadIdx.x;
  if (idx >= KH * KDH * KDH) return;
  int e = idx & 63, d = (idx >> 6) & 63, h = idx >> 12;
  float acc = 0.f;
#pragma unroll
  for (int n = 0; n < KNP; ++n) {
    const float* pn = pers + (size_t)n * KD + h * KDH;
    acc += pn[d] * pn[e];
  }
  mp16[idx] = (_Float16)acc;
}

// ---------------------------------------------------------------------------
// RMSNorm; writes f32 (for gamma MLP) and f16 (GEMM A operand)
// ---------------------------------------------------------------------------
__global__ void rmsnorm_kernel(const float* __restrict__ x,
                               const float* __restrict__ w,
                               float* __restrict__ out,
                               _Float16* __restrict__ out16) {
  __shared__ float red[256];
  int bs = blockIdx.x, t = threadIdx.x;
  const float* xv = x + (size_t)bs * KD;
  float vals[4], ss = 0.f;
#pragma unroll
  for (int j = 0; j < 4; ++j) { float v = xv[t + 256 * j]; vals[j] = v; ss += v * v; }
  red[t] = ss;
  __syncthreads();
  for (int off = 128; off >= 1; off >>= 1) {
    if (t < off) red[t] += red[t + off];
    __syncthreads();
  }
  float r = rsqrtf(red[0] * (1.f / (float)KD) + 1e-6f);
#pragma unroll
  for (int j = 0; j < 4; ++j) {
    int d = t + 256 * j;
    float v = vals[j] * r * w[d];
    out[(size_t)bs * KD + d] = v;
    out16[(size_t)bs * KD + d] = (_Float16)v;
  }
}

// ---------------------------------------------------------------------------
// Gamma gate: g = sigmoid(w2 . silu(w1 @ h)); block = 64 threads per (b,s)
// ---------------------------------------------------------------------------
__global__ void gamma_kernel(const float* __restrict__ h,
                             const float* __restrict__ w1,
                             const float* __restrict__ w2,
                             float* __restrict__ gkout) {
  __shared__ float red[64];
  int bs = blockIdx.x, g = threadIdx.x;
  const float* hv = h + (size_t)bs * KD;
  const float* wr = w1 + (size_t)g * KD;
  float acc = 0.f;
  for (int d = 0; d < KD; ++d) acc += wr[d] * hv[d];
  float s = acc / (1.f + expf(-acc));  // silu
  red[g] = s * w2[g];
  __syncthreads();
  for (int off = 32; off >= 1; off >>= 1) {
    if (g < off) red[g] += red[g + off];
    __syncthreads();
  }
  if (g == 0) gkout[bs] = 1.f / (1.f + expf(-red[0]));
}

// ---------------------------------------------------------------------------
// norm[b][t] = NP + inclusive_cumsum(gk)[t]; one block (1024 thr) per b
// ---------------------------------------------------------------------------
__global__ void cumsum_kernel(const float* __restrict__ gkv, float* __restrict__ normb) {
  __shared__ float buf[KS];
  int t = threadIdx.x, b = blockIdx.x;
  buf[t] = gkv[(size_t)b * KS + t];
  __syncthreads();
  for (int off = 1; off < KS; off <<= 1) {
    float v = (t >= off) ? buf[t - off] : 0.f;
    __syncthreads();
    buf[t] += v;
    __syncthreads();
  }
  normb[(size_t)b * KS + t] = (float)KNP + buf[t];
}

// ---------------------------------------------------------------------------
// qk-norm + RoPE + f16 packing. One wave per (b,s,h); lane handles d, d+32.
// Outputs: q16,k16 (b,h,s,d); kT16,vT16 (b,h,d,s)
// ---------------------------------------------------------------------------
__global__ void qkvprep_kernel(const float* __restrict__ qkv,
                               const float* __restrict__ qn_w,
                               const float* __restrict__ kn_w,
                               const float* __restrict__ cosb,
                               const float* __restrict__ sinb,
                               _Float16* __restrict__ q16,
                               _Float16* __restrict__ k16,
                               _Float16* __restrict__ kT16,
                               _Float16* __restrict__ vT16) {
  int bid = blockIdx.x;
  int h = bid & 15, s = (bid >> 4) & (KS - 1), b = bid >> 14;
  int l = threadIdx.x;
  const float* base = qkv + ((size_t)b * KS + s) * (3 * KD);
  float cs = cosb[s * 32 + l], sn = sinb[s * 32 + l];
  size_t bh = (size_t)b * KH + h;

  { // q
    float x1 = base[h * KDH + l], x2 = base[h * KDH + l + 32];
    float ss = x1 * x1 + x2 * x2;
    for (int off = 16; off >= 1; off >>= 1) ss += __shfl_xor(ss, off, 32);
    float r = rsqrtf(ss * (1.f / (float)KDH) + 1e-6f);
    x1 = x1 * r * qn_w[l];
    x2 = x2 * r * qn_w[l + 32];
    float o1 = x1 * cs - x2 * sn, o2 = x2 * cs + x1 * sn;
    _Float16* qd = q16 + (bh * KS + s) * KDH;
    qd[l] = (_Float16)o1; qd[l + 32] = (_Float16)o2;
  }
  { // k -> k16 and kT16
    float x1 = base[KD + h * KDH + l], x2 = base[KD + h * KDH + l + 32];
    float ss = x1 * x1 + x2 * x2;
    for (int off = 16; off >= 1; off >>= 1) ss += __shfl_xor(ss, off, 32);
    float r = rsqrtf(ss * (1.f / (float)KDH) + 1e-6f);
    x1 = x1 * r * kn_w[l];
    x2 = x2 * r * kn_w[l + 32];
    float o1 = x1 * cs - x2 * sn, o2 = x2 * cs + x1 * sn;
    _Float16* kd = k16 + (bh * KS + s) * KDH;
    kd[l] = (_Float16)o1; kd[l + 32] = (_Float16)o2;
    _Float16* kt = kT16 + bh * KDH * KS;
    kt[(size_t)l * KS + s] = (_Float16)o1;
    kt[(size_t)(l + 32) * KS + s] = (_Float16)o2;
  }
  { // v -> vT16 only
    float v1 = base[2 * KD + h * KDH + l], v2 = base[2 * KD + h * KDH + l + 32];
    _Float16* vt = vT16 + bh * KDH * KS;
    vt[(size_t)l * KS + s] = (_Float16)v1;
    vt[(size_t)(l + 32) * KS + s] = (_Float16)v2;
  }
}

// ---------------------------------------------------------------------------
// Phase 1: Omega-rule memory update. One wave per (b,h,qtile of 16).
// q_new = (1-gate)*q + gate*(Mp q + sum_{k<=q} gk_k (k.q) k) / norm[q]
// ---------------------------------------------------------------------------
__global__ void phase1_kernel(_Float16* __restrict__ q16,
                              const _Float16* __restrict__ k16,
                              const _Float16* __restrict__ kT16,
                              const float* __restrict__ gkb,
                              const float* __restrict__ normb,
                              const _Float16* __restrict__ mp16,
                              const float* __restrict__ memory_gate, int layer) {
  __shared__ __align__(16) _Float16 sS[16 * 40];
  int bid = blockIdx.x;
  int qt = bid & 63, h = (bid >> 6) & 15, b = bid >> 10;
  int l = threadIdx.x & 31;
  int lane15 = l & 15, hi = l >> 4;
  size_t bh = (size_t)b * KH + h;
  _Float16* qp = q16 + bh * KS * KDH;
  const _Float16* kp = k16 + bh * KS * KDH;
  const _Float16* ktp = kT16 + bh * KDH * KS;

  const _Float16* qrow = qp + (size_t)(qt * 16) * KDH;
  v16h qa0 = load_a_frag(qrow + 0, KDH);   // d = 0..31
  v16h qa1 = load_a_frag(qrow + 32, KDH);  // d = 32..63

  v8f macc[4];
#pragma unroll
  for (int dt = 0; dt < 4; ++dt) macc[dt] = zero8();

  int nkc = (qt + 2) >> 1;  // 32-key chunks covering keys 0..qt*16+15
  for (int kc = 0; kc < nkc; ++kc) {
    int s0 = kc * 32;
    v8f st[2];
#pragma unroll
    for (int t = 0; t < 2; ++t) {
      const _Float16* kb = kp + (size_t)(s0 + t * 16) * KDH;
      v16h b0 = load_b_frag(kb + 0, KDH);
      v16h b1 = load_b_frag(kb + 32, KDH);
      v8f s = zero8();
      s = wmma16(qa0, b0, s);
      s = wmma16(qa1, b1, s);
      st[t] = s;
    }
#pragma unroll
    for (int t = 0; t < 2; ++t) {
      int key = s0 + t * 16 + lane15;
      float gkv = gkb[(size_t)b * KS + key];
#pragma unroll
      for (int r = 0; r < 8; ++r) {
        int qr = qt * 16 + hi * 8 + r;
        float v = (key <= qr) ? st[t][r] * gkv : 0.f;
        sS[(hi * 8 + r) * 40 + t * 16 + lane15] = (_Float16)v;
      }
    }
    __syncthreads();
    v16h pa = load_a_frag(sS, 40);
    __syncthreads();
#pragma unroll
    for (int dt = 0; dt < 4; ++dt) {
      v16h bt = load_b_frag(ktp + (size_t)(dt * 16) * KS + s0, KS);
      macc[dt] = wmma16(pa, bt, macc[dt]);
    }
  }
  // mpq: C(q,d) += sum_e Q(q,e) * Mp[d][e]
#pragma unroll
  for (int dt = 0; dt < 4; ++dt) {
    const _Float16* mp = mp16 + ((size_t)h * KDH + dt * 16) * KDH;
    macc[dt] = wmma16(qa0, load_b_frag(mp + 0, KDH), macc[dt]);
    macc[dt] = wmma16(qa1, load_b_frag(mp + 32, KDH), macc[dt]);
  }
  // blend q_new = (1-gate)*q + gate*q_proj
  float gate = 1.f / (1.f + expf(-memory_gate[layer]));
  float nv[8];
#pragma unroll
  for (int r = 0; r < 8; ++r)
    nv[r] = normb[(size_t)b * KS + qt * 16 + hi * 8 + r];
#pragma unroll
  for (int dt = 0; dt < 4; ++dt) {
    int d = dt * 16 + lane15;
#pragma unroll
    for (int r = 0; r < 8; ++r) {
      int sr = qt * 16 + hi * 8 + r;
      _Float16* qe = qp + (size_t)sr * KDH + d;
      float qorig = (float)(*qe);
      float qproj = macc[dt][r] / nv[r];
      *qe = (_Float16)((1.f - gate) * qorig + gate * qproj);
    }
  }
}

// ---------------------------------------------------------------------------
// Phase 2: causal flash attention. One wave per (b,h,qtile of 16).
// ---------------------------------------------------------------------------
__global__ void phase2_kernel(const _Float16* __restrict__ q16,
                              const _Float16* __restrict__ k16,
                              const _Float16* __restrict__ vT16,
                              _Float16* __restrict__ o16) {
  __shared__ __align__(16) _Float16 sS[16 * 40];
  int bid = blockIdx.x;
  int qt = bid & 63, h = (bid >> 6) & 15, b = bid >> 10;
  int l = threadIdx.x & 31;
  int lane15 = l & 15, hi = l >> 4;
  size_t bh = (size_t)b * KH + h;
  const _Float16* qp = q16 + bh * KS * KDH;
  const _Float16* kp = k16 + bh * KS * KDH;
  const _Float16* vtp = vT16 + bh * KDH * KS;
  const float scale = 0.125f;  // 64^-0.5

  const _Float16* qrow = qp + (size_t)(qt * 16) * KDH;
  v16h qa0 = load_a_frag(qrow + 0, KDH);
  v16h qa1 = load_a_frag(qrow + 32, KDH);

  float m_r[8], l_r[8];
  v8f acc[4];
#pragma unroll
  for (int r = 0; r < 8; ++r) { m_r[r] = -3.0e38f; l_r[r] = 0.f; }
#pragma unroll
  for (int dt = 0; dt < 4; ++dt) acc[dt] = zero8();

  int nkc = (qt + 2) >> 1;
  for (int kc = 0; kc < nkc; ++kc) {
    int s0 = kc * 32;
    v8f st[2];
#pragma unroll
    for (int t = 0; t < 2; ++t) {
      const _Float16* kb = kp + (size_t)(s0 + t * 16) * KDH;
      v16h b0 = load_b_frag(kb + 0, KDH);
      v16h b1 = load_b_frag(kb + 32, KDH);
      v8f s = zero8();
      s = wmma16(qa0, b0, s);
      s = wmma16(qa1, b1, s);
      st[t] = s;
    }
#pragma unroll
    for (int t = 0; t < 2; ++t) {
      int key = s0 + t * 16 + lane15;
#pragma unroll
      for (int r = 0; r < 8; ++r) {
        int qr = qt * 16 + hi * 8 + r;
        st[t][r] = (key <= qr) ? st[t][r] * scale : -3.0e38f;
      }
    }
#pragma unroll
    for (int r = 0; r < 8; ++r) {
      float rm = fmaxf(st[0][r], st[1][r]);
      for (int off = 8; off >= 1; off >>= 1) rm = fmaxf(rm, __shfl_xor(rm, off, 16));
      float mnew = fmaxf(m_r[r], rm);
      float corr = expf(m_r[r] - mnew);
      float p0 = expf(st[0][r] - mnew);
      float p1 = expf(st[1][r] - mnew);
      float rs = p0 + p1;
      for (int off = 8; off >= 1; off >>= 1) rs += __shfl_xor(rs, off, 16);
      l_r[r] = l_r[r] * corr + rs;
      m_r[r] = mnew;
#pragma unroll
      for (int dt = 0; dt < 4; ++dt) acc[dt][r] *= corr;
      sS[(hi * 8 + r) * 40 + 0 * 16 + lane15] = (_Float16)p0;
      sS[(hi * 8 + r) * 40 + 1 * 16 + lane15] = (_Float16)p1;
    }
    __syncthreads();
    v16h pa = load_a_frag(sS, 40);
    __syncthreads();
#pragma unroll
    for (int dt = 0; dt < 4; ++dt) {
      v16h bt = load_b_frag(vtp + (size_t)(dt * 16) * KS + s0, KS);
      acc[dt] = wmma16(pa, bt, acc[dt]);
    }
  }
  // store o (B,S,D) f16
#pragma unroll
  for (int dt = 0; dt < 4; ++dt) {
    int d = dt * 16 + lane15;
#pragma unroll
    for (int r = 0; r < 8; ++r) {
      int sr = qt * 16 + hi * 8 + r;
      o16[((size_t)b * KS + sr) * KD + h * KDH + d] = (_Float16)(acc[dt][r] / l_r[r]);
    }
  }
}

// ---------------------------------------------------------------------------
// GEMM: Out(M,N) = A(M,K) @ W(N,K)^T [+ Res]. A,W are f16; Out/Res f32.
// Block 256 thr = 8 waves, tile 64(M) x 128(N), BK=32, wave tile 32x32.
// Fast path (full tile, K%32==0): double-buffered async global->LDS copies.
// ---------------------------------------------------------------------------
template <int ADD_RES>
__global__ __launch_bounds__(256) void gemm_kernel(
    const _Float16* __restrict__ A, int lda,
    const _Float16* __restrict__ W, int ldw,
    const float* __restrict__ Res, float* __restrict__ Out,
    int M, int N, int K) {
  __shared__ __align__(16) _Float16 sA[2][64 * 40];
  __shared__ __align__(32) _Float16 sB[2][128 * 48];
  int tid = threadIdx.x;
  int w = tid >> 5;
  int wm = (w >> 2) * 32;  // 0 or 32
  int wn = (w & 3) * 32;   // 0,32,64,96
  int m0 = blockIdx.y * 64;
  int n0 = blockIdx.x * 128;

  v8f c00 = zero8(), c01 = zero8(), c10 = zero8(), c11 = zero8();

  bool fast = ((K & 31) == 0) && (m0 + 64 <= M) && (n0 + 128 <= N);
#if ASYNC_COPY
  if (fast) {
    int ar = tid >> 2, ac = (tid & 3) * 8;  // A: 64 rows x 32, 8 halfs/thread
    int br = tid >> 1, bc = (tid & 1) * 16; // B: 128 rows x 32, 16 halfs/thread
    auto issue = [&](int buf, int k0) {
      async_cp16(A + (size_t)(m0 + ar) * lda + k0 + ac, &sA[buf][ar * 40 + ac]);
      const _Float16* gb = W + (size_t)(n0 + br) * ldw + k0 + bc;
      _Float16* lb = &sB[buf][br * 48 + bc];
      async_cp16(gb, lb);
      async_cp16(gb + 8, lb + 8);
    };
    issue(0, 0);
    int cur = 0;
    for (int k0 = 0; k0 < K; k0 += 32) {
      wait_async0();      // this wave's async copies into sA/sB[cur] done
      __syncthreads();    // -> whole tile visible to all waves
      if (k0 + 32 < K) issue(cur ^ 1, k0 + 32);  // overlap DMA with WMMA
      v16h a0 = load_a_frag(&sA[cur][(size_t)wm * 40], 40);
      v16h a1 = load_a_frag(&sA[cur][(size_t)(wm + 16) * 40], 40);
      v16h b0 = load_b_frag(&sB[cur][(size_t)wn * 48], 48);
      v16h b1 = load_b_frag(&sB[cur][(size_t)(wn + 16) * 48], 48);
      c00 = wmma16(a0, b0, c00);
      c01 = wmma16(a0, b1, c01);
      c10 = wmma16(a1, b0, c10);
      c11 = wmma16(a1, b1, c11);
      __syncthreads();    // all waves done reading buf[cur] before reuse
      cur ^= 1;
    }
  } else
#endif
  {
    (void)fast;
    for (int k0 = 0; k0 < K; k0 += 32) {
      for (int idx = tid; idx < 64 * 32; idx += 256) {
        int r = idx >> 5, c = idx & 31;
        int gm = m0 + r, gkk = k0 + c;
        sA[0][r * 40 + c] =
            (gm < M && gkk < K) ? A[(size_t)gm * lda + gkk] : (_Float16)0.f;
      }
      for (int idx = tid; idx < 128 * 32; idx += 256) {
        int r = idx >> 5, c = idx & 31;
        int gn = n0 + r, gkk = k0 + c;
        sB[0][r * 48 + c] =
            (gn < N && gkk < K) ? W[(size_t)gn * ldw + gkk] : (_Float16)0.f;
      }
      if (k0 + 32 < K) {  // L2 prefetch of next tiles (global_prefetch_b8)
        __builtin_prefetch(A + (size_t)(m0 + (tid >> 2)) * lda + k0 + 32, 0, 1);
        __builtin_prefetch(W + (size_t)(n0 + (tid >> 1)) * ldw + k0 + 32, 0, 1);
      }
      __syncthreads();
      v16h a0 = load_a_frag(&sA[0][(size_t)wm * 40], 40);
      v16h a1 = load_a_frag(&sA[0][(size_t)(wm + 16) * 40], 40);
      v16h b0 = load_b_frag(&sB[0][(size_t)wn * 48], 48);
      v16h b1 = load_b_frag(&sB[0][(size_t)(wn + 16) * 48], 48);
      c00 = wmma16(a0, b0, c00);
      c01 = wmma16(a0, b1, c01);
      c10 = wmma16(a1, b0, c10);
      c11 = wmma16(a1, b1, c11);
      __syncthreads();
    }
  }

  int l = tid & 31;
  int lane15 = l & 15, hi = l >> 4;
  auto storeC = [&](v8f c, int row0, int col0) {
    int col = col0 + lane15;
    if (col >= N) return;
    int rb = row0 + hi * 8;
#pragma unroll
    for (int r = 0; r < 8; ++r) {
      int row = rb + r;
      if (row < M) {
        float v = c[r];
        if (ADD_RES) v += Res[(size_t)row * N + col];
        Out[(size_t)row * N + col] = v;
      }
    }
  };
  storeC(c00, m0 + wm, n0 + wn);
  storeC(c01, m0 + wm, n0 + wn + 16);
  storeC(c10, m0 + wm + 16, n0 + wn);
  storeC(c11, m0 + wm + 16, n0 + wn + 16);
}

// ---------------------------------------------------------------------------
// SwiGLU: out16[r][c<KFF] = silu(u)*v, pad cols [KFF,KFFP) = 0  (f16, ld=KFFP)
// ---------------------------------------------------------------------------
__global__ void swiglu_kernel(const float* __restrict__ u,
                              const float* __restrict__ vv,
                              _Float16* __restrict__ out16, int rows) {
  long long i = (long long)blockIdx.x * 256 + threadIdx.x;
  long long total = (long long)rows * KFFP;
  if (i >= total) return;
  int c = (int)(i % KFFP);
  long long r = i / KFFP;
  float val = 0.f;
  if (c < KFF) {
    float a = u[r * KFF + c];
    val = (a / (1.f + expf(-a))) * vv[r * KFF + c];
  }
  out16[i] = (_Float16)val;
}

// ---------------------------------------------------------------------------
// Host launch
// ---------------------------------------------------------------------------
extern "C" void kernel_launch(void* const* d_in, const int* in_sizes, int n_in,
                              void* d_out, int out_size, void* d_ws, size_t ws_size,
                              hipStream_t stream) {
  (void)in_sizes; (void)n_in; (void)out_size; (void)ws_size;
  const int*   input_ids  = (const int*)d_in[0];
  const float* tok_emb    = (const float*)d_in[1];
  const float* w_init     = (const float*)d_in[2];
  const float* persistent = (const float*)d_in[3];
  const float* norm1_w    = (const float*)d_in[4];
  const float* norm2_w    = (const float*)d_in[5];
  const float* norm_f_w   = (const float*)d_in[6];
  const float* qkv_w      = (const float*)d_in[7];
  const float* q_norm_w   = (const float*)d_in[8];
  const float* k_norm_w   = (const float*)d_in[9];
  const float* gamma_w1   = (const float*)d_in[10];
  const float* gamma_w2   = (const float*)d_in[11];
  const float* memory_gate= (const float*)d_in[12];
  const float* w_o        = (const float*)d_in[13];
  const float* ffn_w1     = (const float*)d_in[14];
  const float* ffn_w2     = (const float*)d_in[15];
  const float* ffn_w3     = (const float*)d_in[16];
  float* logits = (float*)d_out;

  size_t off = 0;
  auto carve = [&](size_t bytes) -> void* {
    void* p = (char*)d_ws + off;
    off += (bytes + 255) & ~(size_t)255;
    return p;
  };
  const size_t MSD = (size_t)KB * KS;  // 2048 rows
  // activations
  float*    x     = (float*)carve(MSD * KD * 4);
  float*    hb    = (float*)carve(MSD * KD * 4);
  _Float16* hb16  = (_Float16*)carve(MSD * KD * 2);
  float*    qkv   = (float*)carve(MSD * 3 * KD * 4);
  _Float16* ob16  = (_Float16*)carve(MSD * KD * 2);
  float*    ub    = (float*)carve(MSD * KFF * 4);
  float*    vvb   = (float*)carve(MSD * KFF * 4);
  _Float16* ub16  = (_Float16*)carve(MSD * KFFP * 2);
  _Float16* q16   = (_Float16*)carve((size_t)KB * KH * KS * KDH * 2);
  _Float16* k16   = (_Float16*)carve((size_t)KB * KH * KS * KDH * 2);
  _Float16* kT16  = (_Float16*)carve((size_t)KB * KH * KDH * KS * 2);
  _Float16* vT16  = (_Float16*)carve((size_t)KB * KH * KDH * KS * 2);
  _Float16* mp16  = (_Float16*)carve((size_t)KH * KDH * KDH * 2);
  float*    cosb  = (float*)carve((size_t)KS * 32 * 4);
  float*    sinb  = (float*)carve((size_t)KS * 32 * 4);
  float*    gkb   = (float*)carve(MSD * 4);
  float*    nrmb  = (float*)carve(MSD * 4);
  // f16 weight caches
  long long nQKVW = (long long)KL * 3 * KD * KD;
  long long nWO   = (long long)KL * KD * KD;
  long long nW12  = (long long)KL * KFF * KD;
  long long nW3   = (long long)KL * KD * KFFP;
  long long nEMB  = (long long)KV * KD;
  _Float16* qkvw16 = (_Float16*)carve((size_t)nQKVW * 2);
  _Float16* wo16   = (_Float16*)carve((size_t)nWO * 2);
  _Float16* w116   = (_Float16*)carve((size_t)nW12 * 2);
  _Float16* w216   = (_Float16*)carve((size_t)nW12 * 2);
  _Float16* w316   = (_Float16*)carve((size_t)nW3 * 2);
  _Float16* emb16  = (_Float16*)carve((size_t)nEMB * 2);

  auto cvt = [&](const float* s, _Float16* d, long long n) {
    cvt16_kernel<<<(int)((n + 255) / 256), 256, 0, stream>>>(s, d, n);
  };
  cvt(qkv_w, qkvw16, nQKVW);
  cvt(w_o, wo16, nWO);
  cvt(ffn_w1, w116, nW12);
  cvt(ffn_w2, w216, nW12);
  cvtw3_kernel<<<(int)((nW3 + 255) / 256), 256, 0, stream>>>(ffn_w3, w316);
  cvt(tok_emb, emb16, nEMB);

  embed_kernel<<<(int)MSD, 256, 0, stream>>>(input_ids, tok_emb, w_init, x);
  rope_kernel<<<(KS * 32 + 255) / 256, 256, 0, stream>>>(cosb, sinb);
  mp_kernel<<<(KH * KDH * KDH + 255) / 256, 256, 0, stream>>>(persistent, mp16);

  dim3 gQKV((3 * KD + 127) / 128, (int)(MSD / 64));
  dim3 gWO((KD + 127) / 128, (int)(MSD / 64));
  dim3 gFF((KFF + 127) / 128, (int)(MSD / 64));
  dim3 gW3((KD + 127) / 128, (int)(MSD / 64));
  dim3 gLM((KV + 127) / 128, (int)(MSD / 64));
  long long nFFP = (long long)MSD * KFFP;

  for (int i = 0; i < KL; ++i) {
    rmsnorm_kernel<<<(int)MSD, 256, 0, stream>>>(x, norm1_w + (size_t)i * KD, hb, hb16);
    gemm_kernel<0><<<gQKV, 256, 0, stream>>>(hb16, KD,
        qkvw16 + (size_t)i * 3 * KD * KD, KD, nullptr, qkv, (int)MSD, 3 * KD, KD);
    gamma_kernel<<<(int)MSD, 64, 0, stream>>>(hb, gamma_w1 + (size_t)i * KGH * KD,
                                              gamma_w2 + (size_t)i * KGH, gkb);
    cumsum_kernel<<<KB, KS, 0, stream>>>(gkb, nrmb);
    qkvprep_kernel<<<(int)(MSD * KH), 32, 0, stream>>>(
        qkv, q_norm_w + (size_t)i * KDH, k_norm_w + (size_t)i * KDH,
        cosb, sinb, q16, k16, kT16, vT16);
    phase1_kernel<<<KB * KH * (KS / 16), 32, 0, stream>>>(
        q16, k16, kT16, gkb, nrmb, mp16, memory_gate, i);
    phase2_kernel<<<KB * KH * (KS / 16), 32, 0, stream>>>(q16, k16, vT16, ob16);
    gemm_kernel<1><<<gWO, 256, 0, stream>>>(ob16, KD,
        wo16 + (size_t)i * KD * KD, KD, x, x, (int)MSD, KD, KD);
    rmsnorm_kernel<<<(int)MSD, 256, 0, stream>>>(x, norm2_w + (size_t)i * KD, hb, hb16);
    gemm_kernel<0><<<gFF, 256, 0, stream>>>(hb16, KD,
        w116 + (size_t)i * KFF * KD, KD, nullptr, ub, (int)MSD, KFF, KD);
    gemm_kernel<0><<<gFF, 256, 0, stream>>>(hb16, KD,
        w216 + (size_t)i * KFF * KD, KD, nullptr, vvb, (int)MSD, KFF, KD);
    swiglu_kernel<<<(int)((nFFP + 255) / 256), 256, 0, stream>>>(ub, vvb, ub16, (int)MSD);
    gemm_kernel<1><<<gW3, 256, 0, stream>>>(ub16, KFFP,
        w316 + (size_t)i * KD * KFFP, KFFP, x, x, (int)MSD, KD, KFFP);
  }
  rmsnorm_kernel<<<(int)MSD, 256, 0, stream>>>(x, norm_f_w, hb, hb16);
  gemm_kernel<0><<<gLM, 256, 0, stream>>>(hb16, KD, emb16, KD, nullptr, logits,
                                          (int)MSD, KV, KD);
}
